// DeformCNN_67199058313464
// MI455X (gfx1250) — matrix operand order repaired and checked by reading, not compiled, not run on hardware
//
#include <hip/hip_runtime.h>

typedef __attribute__((ext_vector_type(16))) __bf16        v16bf;
typedef __attribute__((ext_vector_type(8)))  float         v8f;
typedef __attribute__((ext_vector_type(8)))  unsigned int  v8u;

static __device__ __forceinline__ unsigned short f32_to_bf16(float f) {
  unsigned int u = __float_as_uint(f);
  u += 0x7fffu + ((u >> 16) & 1u);   // round-to-nearest-even
  return (unsigned short)(u >> 16);
}

// ---------------------------------------------------------------------------
// Offset convolution: off[b, 2K, Ho, Wo] = conv(x, w_off, s=1, pad) + b_off
// ---------------------------------------------------------------------------
__global__ void offconv_kernel(const float* __restrict__ x, const float* __restrict__ w,
                               const float* __restrict__ bias, float* __restrict__ off,
                               int B, int Cin, int H, int W, int K2, int ksz, int pad,
                               int Ho, int Wo) {
  long long t = (long long)blockIdx.x * blockDim.x + threadIdx.x;
  long long total = (long long)B * K2 * Ho * Wo;
  if (t >= total) return;
  int wo = (int)(t % Wo); long long r = t / Wo;
  int ho = (int)(r % Ho); r /= Ho;
  int oc = (int)(r % K2); int b = (int)(r / K2);
  float acc = bias[oc];
  for (int c = 0; c < Cin; ++c) {
    const float* xp = x + (long long)(b * Cin + c) * H * W;
    const float* wp = w + (long long)(oc * Cin + c) * ksz * ksz;
    for (int ky = 0; ky < ksz; ++ky) {
      int iy = ho - pad + ky;
      if (iy < 0 || iy >= H) continue;
      for (int kx = 0; kx < ksz; ++kx) {
        int ix = wo - pad + kx;
        if (ix < 0 || ix >= W) continue;
        acc = fmaf(xp[(long long)iy * W + ix], wp[ky * ksz + kx], acc);
      }
    }
  }
  off[t] = acc;
}

// ---------------------------------------------------------------------------
// Bilinear sampling into WMMA-A-fragment-swizzled bf16 patch buffer.
// Fragment (mt,kt) = 512 bf16 = 32 lanes x 16 elems, stored at
//   patch[((mt*KT + kt)*512) + lane*16 + e]
// 16-bit A layout (16x32): lane<16 -> K in {0..7,16..23}; lane>=16 -> {8..15,24..31}
// ---------------------------------------------------------------------------
__global__ void sample_kernel(const float* __restrict__ x, const float* __restrict__ off,
                              unsigned short* __restrict__ patch,
                              int B, int Cin, int H, int W, int ksz, int pad,
                              int Ho, int Wo, int M, int MT, int KT, int CK) {
  long long t = (long long)blockIdx.x * blockDim.x + threadIdx.x;
  long long total = (long long)MT * KT * 512;
  if (t >= total) return;
  int e    = (int)(t & 15);
  int lane = (int)((t >> 4) & 31);
  long long frag = t >> 9;
  int kt = (int)(frag % KT);
  int mt = (int)(frag / KT);
  int m    = mt * 16 + (lane & 15);
  int half = lane >> 4;
  int klocal = ((e >> 3) << 4) + (half << 3) + (e & 7);
  int kkidx  = kt * 32 + klocal;
  int KK = ksz * ksz;
  unsigned short val = 0;
  if (m < M && kkidx < CK) {
    int c  = kkidx / KK;
    int kr = kkidx - c * KK;
    int ky = kr / ksz, kx = kr - ky * ksz;
    int wo = m % Wo; int r = m / Wo; int ho = r % Ho; int b = r / Ho;
    long long hw = (long long)Ho * Wo;
    long long obase = ((long long)b * KK + kr) * 2 * hw + (long long)ho * Wo + wo;
    float oy = off[obase];
    float ox = off[obase + hw];
    float py = (float)(ho - pad + ky) + oy;
    float px = (float)(wo - pad + kx) + ox;
    float y0f = floorf(py), x0f = floorf(px);
    float ly = py - y0f, lx = px - x0f;
    int y0 = (int)y0f, x0 = (int)x0f;
    const float* xb = x + ((long long)b * Cin + c) * H * W;
    float v00 = 0.f, v01 = 0.f, v10 = 0.f, v11 = 0.f;
    if ((unsigned)y0 < (unsigned)H) {
      if ((unsigned)x0       < (unsigned)W) v00 = xb[(long long)y0 * W + x0];
      if ((unsigned)(x0 + 1) < (unsigned)W) v01 = xb[(long long)y0 * W + x0 + 1];
    }
    if ((unsigned)(y0 + 1) < (unsigned)H) {
      if ((unsigned)x0       < (unsigned)W) v10 = xb[(long long)(y0 + 1) * W + x0];
      if ((unsigned)(x0 + 1) < (unsigned)W) v11 = xb[(long long)(y0 + 1) * W + x0 + 1];
    }
    float s = v00 * (1.f - ly) * (1.f - lx) + v01 * (1.f - ly) * lx
            + v10 * ly * (1.f - lx)         + v11 * ly * lx;
    val = f32_to_bf16(s);
  }
  patch[t] = val;
}

// ---------------------------------------------------------------------------
// Weights -> WMMA-B-fragment-swizzled bf16.  B (32x16): K = (e>=8?16:0)+(lane>=16?8:0)+(e&7),
// N = lane%16.  Fragment (nt,kt) at wb[((nt*KT + kt)*512) + lane*16 + e].
// ---------------------------------------------------------------------------
__global__ void wprep_kernel(const float* __restrict__ w, unsigned short* __restrict__ wb,
                             int O, int CK, int NT, int KT) {
  long long t = (long long)blockIdx.x * blockDim.x + threadIdx.x;
  long long total = (long long)NT * KT * 512;
  if (t >= total) return;
  int e    = (int)(t & 15);
  int lane = (int)((t >> 4) & 31);
  long long frag = t >> 9;
  int kt = (int)(frag % KT);
  int nt = (int)(frag / KT);
  int n = nt * 16 + (lane & 15);
  int kkidx = kt * 32 + ((e >> 3) << 4) + ((lane >> 4) << 3) + (e & 7);
  unsigned short val = 0;
  if (n < O && kkidx < CK) val = f32_to_bf16(w[(long long)n * CK + kkidx]);
  wb[t] = val;
}

// ---------------------------------------------------------------------------
// WMMA GEMM: out[m,n] = sum_kk A[m,kk]*W[n,kk].
// Wave tile 16M x (NSUB*16)N: one A fragment reused across NSUB B fragments.
// Epilogue fuses bias OR batchnorm, then ReLU, scatters to NCHW fp32.
// ---------------------------------------------------------------------------
template <int NSUB>
__global__ __launch_bounds__(256)
void gemm_bf16_kernel(const unsigned short* __restrict__ A,
                      const unsigned short* __restrict__ Bw,
                      float* __restrict__ out,
                      const float* __restrict__ bias,
                      const float* __restrict__ gam, const float* __restrict__ bet,
                      const float* __restrict__ mu,  const float* __restrict__ var,
                      int M, int O, int KT, int Ho, int Wo, int MT) {
  int lane = threadIdx.x & 31;
  int wave = threadIdx.x >> 5;
  int mt = blockIdx.x * 8 + wave;
  if (mt >= MT) return;                 // wave-uniform: EXEC stays all-ones
  int nt0 = blockIdx.y * NSUB;

  const v8u* ap = (const v8u*)A + (long long)mt * KT * 32 + lane;
  const v8u* bp[NSUB];
#pragma unroll
  for (int s = 0; s < NSUB; ++s)
    bp[s] = (const v8u*)Bw + (long long)(nt0 + s) * KT * 32 + lane;

  v8f acc[NSUB];
#pragma unroll
  for (int s = 0; s < NSUB; ++s) acc[s] = (v8f){};

  for (int kt = 0; kt < KT; ++kt) {
    long long o = (long long)kt * 32;
    __builtin_prefetch((const void*)(ap + o + 128), 0, 1);   // stream A ~4 frags ahead
    v16bf a = __builtin_bit_cast(v16bf, ap[o]);
#pragma unroll
    for (int s = 0; s < NSUB; ++s) {
      v16bf b = __builtin_bit_cast(v16bf, bp[s][o]);
      acc[s] = __builtin_amdgcn_wmma_f32_16x16x32_bf16(false, a, false, b,
                                                       (short)0, acc[s], false, false);
    }
  }

  // C/D layout: N = lane%16, M = vgpr + (lane>=16 ? 8 : 0)
  int col  = lane & 15;
  int mloc = (lane >> 4) << 3;
  long long hw = (long long)Ho * Wo;

  float sc[NSUB], sh[NSUB];
#pragma unroll
  for (int s = 0; s < NSUB; ++s) {
    int n = (nt0 + s) * 16 + col;       // O is a multiple of NSUB*16 -> in range
    sc[s] = 1.f; sh[s] = 0.f;
    if (gam) {
      float inv = gam[n] * rsqrtf(var[n] + 1e-5f);
      sc[s] = inv; sh[s] = bet[n] - mu[n] * inv;
    } else if (bias) {
      sh[s] = bias[n];
    }
  }

#pragma unroll
  for (int v = 0; v < 8; ++v) {
    int m = mt * 16 + mloc + v;
    if (m < M) {
      int wo = m % Wo; int r = m / Wo; int ho = r % Ho; int b = r / Ho;
      long long base = ((long long)b * O) * hw + (long long)ho * Wo + wo;
#pragma unroll
      for (int s = 0; s < NSUB; ++s) {
        int n = (nt0 + s) * 16 + col;
        float val = fmaxf(acc[s][v] * sc[s] + sh[s], 0.f);
        out[base + (long long)n * hw] = val;
      }
    }
  }
}

// ---------------------------------------------------------------------------
// Max pooling (2x2 window, variable stride/pad)
// ---------------------------------------------------------------------------
__global__ void maxpool_kernel(const float* __restrict__ in, float* __restrict__ out,
                               int B, int C, int H, int W,
                               int sh, int sw, int pw, int Ho, int Wo) {
  long long t = (long long)blockIdx.x * blockDim.x + threadIdx.x;
  long long total = (long long)B * C * Ho * Wo;
  if (t >= total) return;
  int wo = (int)(t % Wo); long long r = t / Wo;
  int ho = (int)(r % Ho); r /= Ho;
  int c  = (int)(r % C);  int b = (int)(r / C);
  float mval = -INFINITY;
  for (int dy = 0; dy < 2; ++dy) {
    int iy = ho * sh + dy;
    if (iy < 0 || iy >= H) continue;
    for (int dx = 0; dx < 2; ++dx) {
      int ix = wo * sw - pw + dx;
      if (ix < 0 || ix >= W) continue;
      mval = fmaxf(mval, in[(((long long)b * C + c) * H + iy) * W + ix]);
    }
  }
  out[t] = mval;
}

// ---------------------------------------------------------------------------
// Host orchestration
// ---------------------------------------------------------------------------
extern "C" void kernel_launch(void* const* d_in, const int* in_sizes, int n_in,
                              void* d_out, int out_size, void* d_ws, size_t ws_size,
                              hipStream_t stream) {
  static const int CIN[7]   = {1, 64, 128, 256, 256, 512, 512};
  static const int COUT[7]  = {64, 128, 256, 256, 512, 512, 512};
  static const int KSZ[7]   = {3, 3, 3, 3, 3, 3, 2};
  static const int PAD[7]   = {1, 1, 1, 1, 1, 1, 0};
  static const int HASB[7]  = {1, 1, 0, 1, 0, 1, 0};
  static const int HASBN[7] = {0, 0, 1, 0, 1, 0, 1};
  static const int HASP[7]  = {1, 1, 0, 1, 0, 1, 0};
  static const int PSH[7]   = {2, 2, 0, 2, 0, 2, 0};
  static const int PSW[7]   = {2, 2, 0, 1, 0, 1, 0};
  static const int PPW[7]   = {0, 0, 0, 1, 0, 1, 0};

  // Workspace partition (bytes assumed available in d_ws):
  //   buf0/buf1 : 2 x 50,331,648 f32 (largest activation: 8x64x128x768)
  //   patch     : 120,000,000 bf16   (largest swizzled A: ~113.9M elems, L5)
  //   wbuf      : 4,194,304 bf16     (largest swizzled B: ~2.36M elems)
  //   offbuf    : 16,777,216 f32     (largest offsets: 8x18x128x768 = 14.2M)
  char* ws = (char*)d_ws;
  const long long ACT_ELEMS = 50331648LL;
  float* buf0 = (float*)ws;
  float* buf1 = buf0 + ACT_ELEMS;
  unsigned short* patch = (unsigned short*)(buf1 + ACT_ELEMS);
  unsigned short* wbuf  = patch + 120000000LL;
  float* offbuf = (float*)(wbuf + 4194304LL);

  const float* in = (const float*)d_in[0];
  int B = 8, H = 128, W = 768;
  int cur = 1;

  for (int i = 0; i < 7; ++i) {
    const float* w_off = (const float*)d_in[cur++];
    const float* b_off = (const float*)d_in[cur++];
    const float* wgt   = (const float*)d_in[cur++];
    const float *bias = nullptr, *gam = nullptr, *bet = nullptr, *mu = nullptr, *var = nullptr;
    if (HASB[i])  { bias = (const float*)d_in[cur++]; }
    if (HASBN[i]) { gam = (const float*)d_in[cur++]; bet = (const float*)d_in[cur++];
                    mu  = (const float*)d_in[cur++]; var = (const float*)d_in[cur++]; }

    int cin = CIN[i], cout = COUT[i], k = KSZ[i], pad = PAD[i];
    int Ho = H + 2 * pad - k + 1, Wo = W + 2 * pad - k + 1;
    int K2 = 2 * k * k, KK = k * k, CK = cin * KK;
    int M = B * Ho * Wo;
    int MT = (M + 15) / 16, KT = (CK + 31) / 32, NT = cout / 16;

    { long long tot = (long long)B * K2 * Ho * Wo;
      offconv_kernel<<<(unsigned)((tot + 255) / 256), 256, 0, stream>>>(
        in, w_off, b_off, offbuf, B, cin, H, W, K2, k, pad, Ho, Wo); }

    { long long tot = (long long)MT * KT * 512;
      sample_kernel<<<(unsigned)((tot + 255) / 256), 256, 0, stream>>>(
        in, offbuf, patch, B, cin, H, W, k, pad, Ho, Wo, M, MT, KT, CK); }

    { long long tot = (long long)NT * KT * 512;
      wprep_kernel<<<(unsigned)((tot + 255) / 256), 256, 0, stream>>>(
        wgt, wbuf, cout, CK, NT, KT); }

    float* cb = (i == 6) ? (float*)d_out : ((in == buf0) ? buf1 : buf0);
    if (cout % 128 == 0) {
      dim3 grid((unsigned)((MT + 7) / 8), (unsigned)(cout / 128));
      gemm_bf16_kernel<8><<<grid, 256, 0, stream>>>(
        patch, wbuf, cb, bias, gam, bet, mu, var, M, cout, KT, Ho, Wo, MT);
    } else {
      dim3 grid((unsigned)((MT + 7) / 8), (unsigned)(cout / 64));
      gemm_bf16_kernel<4><<<grid, 256, 0, stream>>>(
        patch, wbuf, cb, bias, gam, bet, mu, var, M, cout, KT, Ho, Wo, MT);
    }

    if (HASP[i]) {
      int sh = PSH[i], sw = PSW[i], pw = PPW[i];
      int Hp = (Ho - 2) / sh + 1;
      int Wp = (Wo + 2 * pw - 2) / sw + 1;
      float* pb = (cb == buf0) ? buf1 : buf0;
      long long tot = (long long)B * cout * Hp * Wp;
      maxpool_kernel<<<(unsigned)((tot + 255) / 256), 256, 0, stream>>>(
        cb, pb, B, cout, Ho, Wo, sh, sw, pw, Hp, Wp);
      in = pb; H = Hp; W = Wp;
    } else {
      in = cb; H = Ho; W = Wo;
    }
  }
  (void)in_sizes; (void)n_in; (void)out_size; (void)ws_size;
}
